// DeepRelax_24936580120612
// MI455X (gfx1250) — matrix operand order repaired and checked by reading, not compiled, not run on hardware
//
#include <hip/hip_runtime.h>
#include <math.h>

// ---------------------------------------------------------------------------
// DeepRelax forward for MI455X (gfx1250, wave32, WMMA bf16 w/ f32 accumulate)
//
// GEMMs: v_wmma_f32_16x16x32_bf16, 2x2 tiles per wave (32x32 C per wave),
// guard-free inner loop, K unrolled x2 with disjoint fragment registers so
// the 4-coexec WMMA WAR hazard window (ISA 7.12.1) is filled by the other
// chunk's WMMAs instead of v_nops. A-fragments: 4x global_load_b128 + cvt_pk.
// B-fragments: weights pre-packed once per launch into bf16 in the exact ISA
// 7.12.2 B-matrix lane order -> one aligned 32B vector load per fragment.
// Scatter (segment_sum over random edge targets) uses global f32 atomics.
// ---------------------------------------------------------------------------

typedef __attribute__((ext_vector_type(16))) __bf16 v16bf;
typedef __attribute__((ext_vector_type(8)))  float  v8f;

#define INV_SQRT_2 0.7071067811865475f
#define INV_SQRT_3 0.5773502691896258f
#define INV_SQRT_H 0.0625f            // 1/sqrt(256)
#define LN_10000   9.210340371976184f

__device__ __forceinline__ float ssilu_f(float v) {
    return (v / (1.0f + __expf(-v))) * (1.0f / 0.6f);
}

// ---- A fragment (16x32 bf16): lane holds floats [base+0..7],[base+16..23]
// of its row, base = kk + (lane>=16 ? 8 : 0). Two 32B runs -> 4x b128 + cvt.
__device__ __forceinline__ v16bf load_a_frag(const float* __restrict__ Arow,
                                             int kk, int half) {
    const float4* p = (const float4*)(Arow + kk + half * 8);
    float4 q0 = p[0], q1 = p[1], q2 = p[4], q3 = p[5];
    v16bf a;
    a[0]  = (__bf16)q0.x; a[1]  = (__bf16)q0.y; a[2]  = (__bf16)q0.z; a[3]  = (__bf16)q0.w;
    a[4]  = (__bf16)q1.x; a[5]  = (__bf16)q1.y; a[6]  = (__bf16)q1.z; a[7]  = (__bf16)q1.w;
    a[8]  = (__bf16)q2.x; a[9]  = (__bf16)q2.y; a[10] = (__bf16)q2.z; a[11] = (__bf16)q2.w;
    a[12] = (__bf16)q3.x; a[13] = (__bf16)q3.y; a[14] = (__bf16)q3.z; a[15] = (__bf16)q3.w;
    return a;
}

// ---- B fragment from packed weights: pack[((kc*NT + nt)*32 + lane)*16 + t]
// where element t of the v16bf is W[kc*32 + half*8 + t + (t>=8 ? 8 : 0), nt*16 + (lane&15)].
// One aligned 32B vector load per fragment.
__device__ __forceinline__ v16bf load_b_packed(const __bf16* __restrict__ pack,
                                               int NT, int kk, int nt, int lane) {
    return *(const v16bf*)(pack + ((((size_t)(kk >> 5)) * NT + nt) * 32 + lane) * 16);
}

// ---- weight repack: f32 [K,N] row-major -> bf16 B-fragment tiles -----------
__global__ void pack_w_kernel(const float* __restrict__ W, __bf16* __restrict__ pack,
                              int K, int N, int NT) {
    int KC = K >> 5;
    int idx = blockIdx.x * blockDim.x + threadIdx.x;   // (kc*NT + nt)*32 + lane
    if (idx >= KC * NT * 32) return;
    int lane = idx & 31;
    int tile = idx >> 5;
    int nt = tile % NT, kc = tile / NT;
    int half = lane >> 4, ncol = lane & 15;
    int n = nt * 16 + ncol; n = n < N ? n : N - 1;     // clamp (dup cols unused)
    const float* src = W + (size_t)(kc * 32 + half * 8) * N + n;
    __bf16* dst = pack + (size_t)idx * 16;
#pragma unroll
    for (int t = 0; t < 16; ++t) dst[t] = (__bf16)src[(size_t)(t + (t >= 8 ? 8 : 0)) * N];
}

#define WMMA_BF16(a, b, c) \
    __builtin_amdgcn_wmma_f32_16x16x32_bf16(false, (a), false, (b), (short)0, (c), false, false)

// ---- Generic GEMM: C = act(gather(A)[M,K] @ W[K,N] + bias), K%64==0 --------
// 128 threads = 4 waves; each wave owns a 32x32 C tile (2x2 WMMA tiles).
template <int ACT>
__global__ void gemm_kernel(const float* __restrict__ A,
                            const int* __restrict__ rowIdx,
                            const __bf16* __restrict__ Wp, int NT,
                            const float* __restrict__ bias,
                            float* __restrict__ C, int M, int K, int N) {
    int lane  = threadIdx.x & 31;
    int wave  = threadIdx.x >> 5;
    int half  = lane >> 4;
    int nbase = blockIdx.x * 32;
    int mbase = (blockIdx.y * 4 + wave) * 32;
    if (mbase >= M) return;                 // whole-wave uniform exit

    int r0 = mbase + (lane & 15);      r0 = r0 < M ? r0 : M - 1;
    int r1 = mbase + 16 + (lane & 15); r1 = r1 < M ? r1 : M - 1;
    int a0 = rowIdx ? rowIdx[r0] : r0;
    int a1 = rowIdx ? rowIdx[r1] : r1;
    const float* A0 = A + (size_t)a0 * K;
    const float* A1 = A + (size_t)a1 * K;
    int ncol = lane & 15;
    int n0 = nbase + ncol;      int n0c = n0 < N ? n0 : N - 1;
    int n1 = nbase + 16 + ncol; int n1c = n1 < N ? n1 : N - 1;
    int nt0 = blockIdx.x * 2;
    int nt1 = nt0 + 1 < NT ? nt0 + 1 : NT - 1;

    v8f acc00 = {}, acc01 = {}, acc10 = {}, acc11 = {};
    for (int kk = 0; kk < K; kk += 64) {    // K/32 even at every call site
        v16bf af0a = load_a_frag(A0, kk, half);
        v16bf af1a = load_a_frag(A1, kk, half);
        v16bf bf0a = load_b_packed(Wp, NT, kk, nt0, lane);
        v16bf bf1a = load_b_packed(Wp, NT, kk, nt1, lane);
        v16bf af0b = load_a_frag(A0, kk + 32, half);
        v16bf af1b = load_a_frag(A1, kk + 32, half);
        v16bf bf0b = load_b_packed(Wp, NT, kk + 32, nt0, lane);
        v16bf bf1b = load_b_packed(Wp, NT, kk + 32, nt1, lane);
        acc00 = WMMA_BF16(af0a, bf0a, acc00);
        acc01 = WMMA_BF16(af0a, bf1a, acc01);
        acc10 = WMMA_BF16(af1a, bf0a, acc10);
        acc11 = WMMA_BF16(af1a, bf1a, acc11);
        acc00 = WMMA_BF16(af0b, bf0b, acc00);
        acc01 = WMMA_BF16(af0b, bf1b, acc01);
        acc10 = WMMA_BF16(af1b, bf0b, acc10);
        acc11 = WMMA_BF16(af1b, bf1b, acc11);
    }
    float b0 = bias ? bias[n0c] : 0.0f;
    float b1 = bias ? bias[n1c] : 0.0f;
#pragma unroll
    for (int r = 0; r < 8; ++r) {
        int m0 = mbase + half * 8 + r;
        int m1 = m0 + 16;
        float v00 = acc00[r] + b0, v01 = acc01[r] + b1;
        float v10 = acc10[r] + b0, v11 = acc11[r] + b1;
        if (ACT == 1) { v00 = ssilu_f(v00); v01 = ssilu_f(v01);
                        v10 = ssilu_f(v10); v11 = ssilu_f(v11); }
        if (m0 < M) {
            if (n0 < N) C[(size_t)m0 * N + n0] = v00;
            if (n1 < N) C[(size_t)m0 * N + n1] = v01;
        }
        if (m1 < M) {
            if (n0 < N) C[(size_t)m1 * N + n0] = v10;
            if (n1 < N) C[(size_t)m1 * N + n1] = v11;
        }
    }
}

// ---- one 256-wide K segment of the concat-gather GEMM (8 chunks, unroll 2) --
__device__ __forceinline__ void accum_seg(const float* __restrict__ p0,
                                          const float* __restrict__ p1,
                                          const __bf16* __restrict__ Wp, int NT,
                                          int nt0, int nt1, int segbase,
                                          int half, int lane,
                                          v8f& a00, v8f& a01, v8f& a10, v8f& a11) {
    for (int kloc = 0; kloc < 256; kloc += 64) {
        v16bf af0a = load_a_frag(p0, kloc, half);
        v16bf af1a = load_a_frag(p1, kloc, half);
        v16bf bf0a = load_b_packed(Wp, NT, segbase + kloc, nt0, lane);
        v16bf bf1a = load_b_packed(Wp, NT, segbase + kloc, nt1, lane);
        v16bf af0b = load_a_frag(p0, kloc + 32, half);
        v16bf af1b = load_a_frag(p1, kloc + 32, half);
        v16bf bf0b = load_b_packed(Wp, NT, segbase + kloc + 32, nt0, lane);
        v16bf bf1b = load_b_packed(Wp, NT, segbase + kloc + 32, nt1, lane);
        a00 = WMMA_BF16(af0a, bf0a, a00);
        a01 = WMMA_BF16(af0a, bf1a, a01);
        a10 = WMMA_BF16(af1a, bf0a, a10);
        a11 = WMMA_BF16(af1a, bf1a, a11);
        a00 = WMMA_BF16(af0b, bf0b, a00);
        a01 = WMMA_BF16(af0b, bf1b, a01);
        a10 = WMMA_BF16(af1b, bf0b, a10);
        a11 = WMMA_BF16(af1b, bf1b, a11);
    }
}

// ---- Concat-gather GEMM: row m of A = concat([xf[i], xf[j], ef[m]]), K=768 --
template <int ACT>
__global__ void gemm_feat_kernel(const float* __restrict__ xf,   // [Nnodes,256]
                                 const float* __restrict__ ef,   // [M,256]
                                 const int* __restrict__ eidx,   // [2,E] (j,i)
                                 const int* __restrict__ maskIdx,// optional [M]
                                 int E,
                                 const __bf16* __restrict__ Wp, int NT,
                                 const float* __restrict__ bias,
                                 float* __restrict__ C, int M, int N) {
    int lane  = threadIdx.x & 31;
    int wave  = threadIdx.x >> 5;
    int half  = lane >> 4;
    int nbase = blockIdx.x * 32;
    int mbase = (blockIdx.y * 4 + wave) * 32;
    if (mbase >= M) return;

    int r0 = mbase + (lane & 15);      r0 = r0 < M ? r0 : M - 1;
    int r1 = mbase + 16 + (lane & 15); r1 = r1 < M ? r1 : M - 1;
    int e0 = maskIdx ? maskIdx[r0] : r0;
    int e1 = maskIdx ? maskIdx[r1] : r1;
    int ncol = lane & 15;
    int n0 = nbase + ncol;      int n0c = n0 < N ? n0 : N - 1;
    int n1 = nbase + 16 + ncol; int n1c = n1 < N ? n1 : N - 1;
    int nt0 = blockIdx.x * 2;
    int nt1 = nt0 + 1 < NT ? nt0 + 1 : NT - 1;

    v8f acc00 = {}, acc01 = {}, acc10 = {}, acc11 = {};
    accum_seg(xf + (size_t)eidx[E + e0] * 256, xf + (size_t)eidx[E + e1] * 256,
              Wp, NT, nt0, nt1, 0,   half, lane, acc00, acc01, acc10, acc11);
    accum_seg(xf + (size_t)eidx[e0] * 256,     xf + (size_t)eidx[e1] * 256,
              Wp, NT, nt0, nt1, 256, half, lane, acc00, acc01, acc10, acc11);
    accum_seg(ef + (size_t)r0 * 256,           ef + (size_t)r1 * 256,
              Wp, NT, nt0, nt1, 512, half, lane, acc00, acc01, acc10, acc11);

    float b0 = bias ? bias[n0c] : 0.0f;
    float b1 = bias ? bias[n1c] : 0.0f;
#pragma unroll
    for (int r = 0; r < 8; ++r) {
        int m0 = mbase + half * 8 + r;
        int m1 = m0 + 16;
        float v00 = acc00[r] + b0, v01 = acc01[r] + b1;
        float v10 = acc10[r] + b0, v11 = acc11[r] + b1;
        if (ACT == 1) { v00 = ssilu_f(v00); v01 = ssilu_f(v01);
                        v10 = ssilu_f(v10); v11 = ssilu_f(v11); }
        if (m0 < M) {
            if (n0 < N) C[(size_t)m0 * N + n0] = v00;
            if (n1 < N) C[(size_t)m0 * N + n1] = v01;
        }
        if (m1 < M) {
            if (n0 < N) C[(size_t)m1 * N + n0] = v10;
            if (n1 < N) C[(size_t)m1 * N + n1] = v11;
        }
    }
}

// ---- Tiny ragged-K GEMM (lin_cell[0]: 200x9x256, negligible FLOPs) ---------
__global__ void small_gemm_kernel(const float* __restrict__ A, const float* __restrict__ W,
                                  const float* __restrict__ bias, float* __restrict__ C,
                                  int M, int K, int N, int act) {
    int idx = blockIdx.x * blockDim.x + threadIdx.x;
    if (idx >= M * N) return;
    int m = idx / N, n = idx % N;
    float s = bias ? bias[n] : 0.0f;
    for (int k = 0; k < K; ++k) s += A[(size_t)m * K + k] * W[(size_t)k * N + n];
    if (act) s = ssilu_f(s);
    C[idx] = s;
}

// ---- graph id per edge ------------------------------------------------------
__global__ void cum_kernel(const int* __restrict__ nb, int* __restrict__ cum, int G) {
    if (blockIdx.x == 0 && threadIdx.x == 0) {
        int s = 0;
        for (int g = 0; g < G; ++g) { cum[g] = s; s += nb[g]; }
        cum[G] = s;
    }
}

__global__ void egraph_kernel(const int* __restrict__ cum, int* __restrict__ eg,
                              int E, int G) {
    int e = blockIdx.x * blockDim.x + threadIdx.x;
    if (e >= E) return;
    int lo = 0, hi = G - 1;
    while (lo < hi) {
        int mid = (lo + hi + 1) >> 1;
        if (cum[mid] <= e) lo = mid; else hi = mid - 1;
    }
    eg[e] = lo;
}

// ---- edge features: RBF(128) ++ sinusoidal PE(128), plus edge_vector -------
__global__ void edgefeat_kernel(const float* __restrict__ pos,
                                const float* __restrict__ cell,
                                const int* __restrict__ coff,
                                const int* __restrict__ eidx,
                                const int* __restrict__ eg,
                                float* __restrict__ edge_feat,
                                float* __restrict__ evec, int E) {
    size_t idx = (size_t)blockIdx.x * blockDim.x + threadIdx.x;
    if (idx >= (size_t)E * 256) return;
    int e = (int)(idx >> 8), f = (int)(idx & 255);
    int j = eidx[e], i = eidx[E + e];
    int c0 = coff[3 * e], c1 = coff[3 * e + 1], c2 = coff[3 * e + 2];
    const float* cl = cell + (size_t)eg[e] * 9;
    float v[3]; float d2 = 0.0f;
#pragma unroll
    for (int l = 0; l < 3; ++l) {
        float t = pos[3 * j + l] + (float)c0 * cl[l] + (float)c1 * cl[3 + l]
                + (float)c2 * cl[6 + l] - pos[3 * i + l];
        v[l] = t; d2 += t * t;
    }
    float d = sqrtf(d2);
    if (f == 0) {
        float inv = -1.0f / d;
        evec[3 * e] = v[0] * inv; evec[3 * e + 1] = v[1] * inv; evec[3 * e + 2] = v[2] * inv;
    }
    float out;
    if (f < 128) {
        float ds = d * (1.0f / 6.0f);
        float ds2 = ds * ds;
        float ds5 = ds2 * ds2 * ds;
        float env = 1.0f + ds5 * (-21.0f + ds * (35.0f - 15.0f * ds));
        env = (ds < 1.0f) ? env : 0.0f;
        float delta = ds - (float)f * (1.0f / 127.0f);
        out = env * __expf(-8064.5f * delta * delta);
    } else {
        int m = f - 128;
        int p = m >> 1;
        float co_num = (float)((c0 + 1) * 9 + (c1 + 1) * 3 + (c2 + 1));
        float freq = __expf(-LN_10000 * (float)p * (1.0f / 64.0f));
        float ang = co_num * freq;
        out = (m & 1) ? __cosf(ang) : __sinf(ang);
    }
    edge_feat[idx] = out;
}

__global__ void xinit_kernel(const float* __restrict__ emb, const int* __restrict__ z,
                             float* __restrict__ x, int N) {
    size_t idx = (size_t)blockIdx.x * blockDim.x + threadIdx.x;
    if (idx >= (size_t)N * 256) return;
    int n = (int)(idx >> 8), h = (int)(idx & 255);
    x[idx] = emb[(size_t)(z[n] - 1) * 256 + h];
}

// ---- message pass: scatter-add into d_vec / d_x -----------------------------
__global__ void message_kernel(const float* __restrict__ xh,   // [N,768]
                               const float* __restrict__ rbf,  // [E,768]
                               const float* __restrict__ vec,  // [N,3,256]
                               const float* __restrict__ evec, // [E,3]
                               const int* __restrict__ eidx,
                               float* __restrict__ dvec, float* __restrict__ dx,
                               int E) {
    size_t idx = (size_t)blockIdx.x * blockDim.x + threadIdx.x;
    if (idx >= (size_t)E * 256) return;
    int e = (int)(idx >> 8), h = (int)(idx & 255);
    int j = eidx[e], i = eidx[E + e];
    const float* xj = xh + (size_t)j * 768;
    const float* re = rbf + (size_t)e * 768;
    float x1 = xj[h]       * re[h]       * INV_SQRT_3;
    float x2 = xj[256 + h] * re[256 + h] * INV_SQRT_3;
    float x3 = xj[512 + h] * re[512 + h] * INV_SQRT_3;
#pragma unroll
    for (int d = 0; d < 3; ++d) {
        float vji = (x1 * vec[(size_t)j * 768 + d * 256 + h]
                   + x2 * evec[3 * e + d]) * INV_SQRT_H;
        atomicAdd(&dvec[(size_t)i * 768 + d * 256 + h], vji);
    }
    atomicAdd(&dx[(size_t)i * 256 + h], x3);
}

__global__ void vec_add_kernel(float* __restrict__ a, const float* __restrict__ b, size_t n) {
    size_t idx = (size_t)blockIdx.x * blockDim.x + threadIdx.x;
    if (idx < n) a[idx] += b[idx];
}

__global__ void x_update_kernel(float* __restrict__ x, const float* __restrict__ dx, size_t n) {
    size_t idx = (size_t)blockIdx.x * blockDim.x + threadIdx.x;
    if (idx < n) x[idx] = (x[idx] + dx[idx]) * INV_SQRT_2;
}

// ---- vec1.vec2 dot / ||vec2|| and [x, vnorm] assembly -----------------------
__global__ void vecdot_kernel(const float* __restrict__ v12, const float* __restrict__ x,
                              float* __restrict__ vdot, float* __restrict__ xvin, int N) {
    size_t idx = (size_t)blockIdx.x * blockDim.x + threadIdx.x;
    if (idx >= (size_t)N * 256) return;
    int n = (int)(idx >> 8), g = (int)(idx & 255);
    const float* base = v12 + (size_t)n * 1536;
    float dot = 0.0f, nn = 0.0f;
#pragma unroll
    for (int d = 0; d < 3; ++d) {
        float a = base[d * 512 + g];
        float b = base[d * 512 + 256 + g];
        dot += a * b; nn += b * b;
    }
    vdot[idx] = dot * INV_SQRT_H;
    xvin[(size_t)n * 512 + g]       = x[idx];
    xvin[(size_t)n * 512 + 256 + g] = sqrtf(nn + 1e-8f);
}

__global__ void update2_kernel(float* __restrict__ x, float* __restrict__ vec,
                               const float* __restrict__ xvh, const float* __restrict__ vdot,
                               const float* __restrict__ v12, int N) {
    size_t idx = (size_t)blockIdx.x * blockDim.x + threadIdx.x;
    if (idx >= (size_t)N * 256) return;
    int n = (int)(idx >> 8), g = (int)(idx & 255);
    const float* xr = xvh + (size_t)n * 768;
    float xv1 = xr[g], xv2 = xr[256 + g], xv3 = xr[512 + g];
    x[idx] += (xv1 + xv2 * vdot[idx]) * INV_SQRT_2;
    const float* vb = v12 + (size_t)n * 1536;
#pragma unroll
    for (int d = 0; d < 3; ++d)
        vec[(size_t)n * 768 + d * 256 + g] += xv3 * vb[d * 512 + g];
}

// ---- per-graph pooling + output assembly ------------------------------------
__global__ void scatter_g_kernel(const float* __restrict__ xc, const int* __restrict__ batch,
                                 float* __restrict__ g, int N) {
    size_t idx = (size_t)blockIdx.x * blockDim.x + threadIdx.x;
    if (idx >= (size_t)N * 256) return;
    int n = (int)(idx >> 8), h = (int)(idx & 255);
    atomicAdd(&g[(size_t)batch[n] * 256 + h], xc[idx]);
}

__global__ void concat_gc_kernel(const float* __restrict__ g, const float* __restrict__ cf,
                                 float* __restrict__ gc, int G) {
    size_t idx = (size_t)blockIdx.x * blockDim.x + threadIdx.x;
    if (idx >= (size_t)G * 512) return;
    int gi = (int)(idx >> 9), f = (int)(idx & 511);
    gc[idx] = (f < 256) ? g[(size_t)gi * 256 + f] : cf[(size_t)gi * 256 + (f - 256)];
}

__global__ void split_out_kernel(const float* __restrict__ o2, float* __restrict__ out0,
                                 float* __restrict__ out1, int M) {
    int m = blockIdx.x * blockDim.x + threadIdx.x;
    if (m >= M) return;
    out0[m] = fmaxf(o2[2 * m], 0.0f);
    out1[m] = o2[2 * m + 1];
}

__global__ void cell_final_kernel(const float* __restrict__ o9, const float* __restrict__ cellu,
                                  float* __restrict__ out, int G) {
    int idx = blockIdx.x * blockDim.x + threadIdx.x;
    if (idx >= G * 9) return;
    out[idx] = o9[idx] + cellu[idx];
}

// ---------------------------------------------------------------------------
static inline void launch_gemm(hipStream_t s, const float* A, const int* rIdx,
                               const __bf16* Wp, const float* b, float* C,
                               int M, int K, int N, int act) {
    int NT = (N + 15) / 16;
    dim3 gr((N + 31) / 32, (M + 127) / 128), bl(128);
    if (act) gemm_kernel<1><<<gr, bl, 0, s>>>(A, rIdx, Wp, NT, b, C, M, K, N);
    else     gemm_kernel<0><<<gr, bl, 0, s>>>(A, rIdx, Wp, NT, b, C, M, K, N);
}

static inline void launch_gemm_feat(hipStream_t s, const float* xf, const float* ef,
                                    const int* eidx, const int* mask, int E,
                                    const __bf16* Wp, const float* b, float* C,
                                    int M, int N, int act) {
    int NT = (N + 15) / 16;
    dim3 gr((N + 31) / 32, (M + 127) / 128), bl(128);
    if (act) gemm_feat_kernel<1><<<gr, bl, 0, s>>>(xf, ef, eidx, mask, E, Wp, NT, b, C, M, N);
    else     gemm_feat_kernel<0><<<gr, bl, 0, s>>>(xf, ef, eidx, mask, E, Wp, NT, b, C, M, N);
}

#define EW(count) dim3(((size_t)(count) + 255) / 256), dim3(256)

extern "C" void kernel_launch(void* const* d_in, const int* in_sizes, int n_in,
                              void* d_out, int out_size, void* d_ws, size_t ws_size,
                              hipStream_t stream) {
    const float* pos   = (const float*)d_in[0];
    const float* cellu = (const float*)d_in[1];
    const int*   coff  = (const int*)d_in[2];
    const int*   eidx  = (const int*)d_in[3];
    const int*   nbrs  = (const int*)d_in[4];
    const int*   batch = (const int*)d_in[5];
    const int*   z     = (const int*)d_in[6];
    const int*   mask  = (const int*)d_in[7];
    const int N  = in_sizes[0] / 3;
    const int E  = in_sizes[3] / 2;
    const int G  = in_sizes[4];
    const int Mm = in_sizes[7];

    // params pytree (JAX sorted-key order) starting at d_in[8]:
    //  8 atom_emb | 9/10 cell_br | 11/12 dist_disp | 13/14 dist_rel
    //  15..18 lin_cell | 19..22 lin_edge_disp | 23..26 lin_edge_rel
    //  27..44 mp[l]{edge.W,edge.b,x1.W,x1.b,x2.W,x2.b}
    //  45..50 out_cell | 51..56 out_dist_disp | 57..62 out_dist_rel
    //  63..77 up[l]{vec.W,xv1.W,xv1.b,xv2.W,xv2.b}
#define P(k) ((const float*)d_in[(k)])
    const float* atom_emb = P(8);

    // ---- workspace layout (floats), aliased; every region 256B-aligned ----
    float* ws = (float*)d_ws;
    size_t off = 0;
    auto alloc = [&](size_t n) {
        n = (n + 63) & ~(size_t)63;
        float* p = ws + off; off += n; return p;
    };
    int*   egraph = (int*)alloc(E);
    int*   cum    = (int*)alloc(G + 1);
    float* efeat  = alloc((size_t)E * 256);
    float* evec   = alloc((size_t)E * 3);
    float* x      = alloc((size_t)N * 256);
    float* vec    = alloc((size_t)N * 768);
    float* big    = alloc((size_t)E * 768);  // rbf_h | hd1 | hr1
    float* xh     = alloc((size_t)N * 768);  // x_h | xvh | b1d | hd2
    float* dvec   = alloc((size_t)N * 768);  // d_vec | ef_d
    float* dx     = alloc((size_t)N * 256);  // d_x | xdd
    float* v12    = alloc((size_t)N * 1536); // v12 | b1r | hr2
    float* vdot   = alloc((size_t)N * 256);  // vec_dot | xdr
    float* xvin   = alloc((size_t)N * 512);
    float* t1     = alloc((size_t)N * 128);  // t1 | odisp
    float* t2     = alloc((size_t)N * 256);  // t2 | xc
    float* s2buf  = alloc((size_t)E * 256);  // ef_r
    float* gbuf   = alloc((size_t)G * 256);
    float* cf1    = alloc((size_t)G * 256);
    float* cf2    = alloc((size_t)G * 256);
    float* gc     = alloc((size_t)G * 512);
    float* o9     = alloc((size_t)G * 9);
    float* orel   = alloc((size_t)E * 2);
    (void)ws_size; (void)n_in; (void)out_size;

    // ---- pre-pack all WMMA weights to bf16 B-fragment layout (L2-hot) ----
    auto pack_w = [&](const float* W, int K, int N_) -> const __bf16* {
        int KC = K / 32, NT = (N_ + 15) / 16;
        size_t elems = (size_t)KC * NT * 512;            // bf16 count
        __bf16* p = (__bf16*)alloc((elems + 1) / 2);     // alloc unit = float
        pack_w_kernel<<<EW(KC * NT * 32), 0, stream>>>(W, p, K, N_, NT);
        return p;
    };
    const __bf16 *Wedge[3], *Wx1[3], *Wx2[3], *Wvec[3], *Wxv1[3], *Wxv2[3];
    for (int l = 0; l < 3; ++l) {
        Wedge[l] = pack_w(P(27 + 6 * l), 256, 768);
        Wx1[l]   = pack_w(P(29 + 6 * l), 256, 128);
        Wx2[l]   = pack_w(P(31 + 6 * l), 128, 768);
        Wvec[l]  = pack_w(P(63 + 5 * l), 256, 512);
        Wxv1[l]  = pack_w(P(64 + 5 * l), 512, 256);
        Wxv2[l]  = pack_w(P(66 + 5 * l), 256, 768);
    }
    const __bf16* Wcb   = pack_w(P(9),  256, 256);
    const __bf16* Wdd   = pack_w(P(11), 256, 256);
    const __bf16* Wdr   = pack_w(P(13), 256, 256);
    const __bf16* Wlc1  = pack_w(P(17), 256, 256);
    const __bf16* Wled0 = pack_w(P(19), 256, 256);
    const __bf16* Wled1 = pack_w(P(21), 256, 256);
    const __bf16* Wler0 = pack_w(P(23), 256, 256);
    const __bf16* Wler1 = pack_w(P(25), 256, 256);
    const __bf16* Woc0  = pack_w(P(45), 512, 256);
    const __bf16* Woc1  = pack_w(P(47), 256, 256);
    const __bf16* Woc2  = pack_w(P(49), 256, 9);
    const __bf16* Wodd0 = pack_w(P(51), 768, 256);
    const __bf16* Wodd1 = pack_w(P(53), 256, 256);
    const __bf16* Wodd2 = pack_w(P(55), 256, 2);
    const __bf16* Wodr0 = pack_w(P(57), 768, 256);
    const __bf16* Wodr1 = pack_w(P(59), 256, 256);
    const __bf16* Wodr2 = pack_w(P(61), 256, 2);

    float* outD0 = (float*)d_out;
    float* outD1 = outD0 + Mm;
    float* outR0 = outD0 + 2 * (size_t)Mm;
    float* outR1 = outR0 + E;
    float* outC  = outR0 + 2 * (size_t)E;

    // ---- stage 0: edge features, x/vec init ----
    hipMemsetAsync(vec, 0, (size_t)N * 768 * sizeof(float), stream);
    cum_kernel<<<1, 1, 0, stream>>>(nbrs, cum, G);
    egraph_kernel<<<EW(E), 0, stream>>>(cum, egraph, E, G);
    edgefeat_kernel<<<EW((size_t)E * 256), 0, stream>>>(pos, cellu, coff, eidx, egraph,
                                                        efeat, evec, E);
    xinit_kernel<<<EW((size_t)N * 256), 0, stream>>>(atom_emb, z, x, N);

    // ---- message-passing layers ----
    for (int l = 0; l < 3; ++l) {
        const int mb = 27 + 6 * l, ub = 63 + 5 * l;
        launch_gemm(stream, efeat, nullptr, Wedge[l], P(mb + 1), big, E, 256, 768, 0); // rbf_h
        launch_gemm(stream, x,     nullptr, Wx1[l],   P(mb + 3), t1,  N, 256, 128, 1);
        launch_gemm(stream, t1,    nullptr, Wx2[l],   P(mb + 5), xh,  N, 128, 768, 0); // x_h
        hipMemsetAsync(dvec, 0, (size_t)N * 768 * sizeof(float), stream);
        hipMemsetAsync(dx,   0, (size_t)N * 256 * sizeof(float), stream);
        message_kernel<<<EW((size_t)E * 256), 0, stream>>>(xh, big, vec, evec, eidx,
                                                           dvec, dx, E);
        vec_add_kernel<<<EW((size_t)N * 768), 0, stream>>>(vec, dvec, (size_t)N * 768);
        x_update_kernel<<<EW((size_t)N * 256), 0, stream>>>(x, dx, (size_t)N * 256);
        launch_gemm(stream, vec,  nullptr, Wvec[l],  nullptr,   v12, 3 * N, 256, 512, 0);
        vecdot_kernel<<<EW((size_t)N * 256), 0, stream>>>(v12, x, vdot, xvin, N);
        launch_gemm(stream, xvin, nullptr, Wxv1[l],  P(ub + 2), t2,  N, 512, 256, 1);
        launch_gemm(stream, t2,   nullptr, Wxv2[l],  P(ub + 4), xh,  N, 256, 768, 0); // xvh
        update2_kernel<<<EW((size_t)N * 256), 0, stream>>>(x, vec, xh, vdot, v12, N);
    }

    // ---- disp head (masked edges) ----
    launch_gemm(stream, efeat, mask,    Wled0, P(20), xh,   Mm, 256, 256, 1); // b1d
    launch_gemm(stream, xh,    nullptr, Wled1, P(22), dvec, Mm, 256, 256, 1); // ef_d
    launch_gemm(stream, x,     nullptr, Wdd,   P(12), dx,   N,  256, 256, 1); // xdd
    launch_gemm_feat(stream, dx, dvec, eidx, mask, E, Wodd0, P(52), big, Mm, 256, 1); // hd1
    launch_gemm(stream, big,   nullptr, Wodd1, P(54), xh,   Mm, 256, 256, 1); // hd2
    launch_gemm(stream, xh,    nullptr, Wodd2, P(56), t1,   Mm, 256, 2,   0); // od
    split_out_kernel<<<EW(Mm), 0, stream>>>(t1, outD0, outD1, Mm);

    // ---- rel head (all edges) ----
    launch_gemm(stream, efeat, nullptr, Wler0, P(24), v12,   E, 256, 256, 1); // b1r
    launch_gemm(stream, v12,   nullptr, Wler1, P(26), s2buf, E, 256, 256, 1); // ef_r
    launch_gemm(stream, x,     nullptr, Wdr,   P(14), vdot,  N, 256, 256, 1); // xdr
    launch_gemm_feat(stream, vdot, s2buf, eidx, nullptr, E, Wodr0, P(58), big, E, 256, 1); // hr1
    launch_gemm(stream, big,   nullptr, Wodr1, P(60), v12,  E, 256, 256, 1);  // hr2
    launch_gemm(stream, v12,   nullptr, Wodr2, P(62), orel, E, 256, 2,   0);
    split_out_kernel<<<EW(E), 0, stream>>>(orel, outR0, outR1, E);

    // ---- cell head ----
    launch_gemm(stream, x, nullptr, Wcb, P(10), t2, N, 256, 256, 1);           // xc
    hipMemsetAsync(gbuf, 0, (size_t)G * 256 * sizeof(float), stream);
    scatter_g_kernel<<<EW((size_t)N * 256), 0, stream>>>(t2, batch, gbuf, N);
    small_gemm_kernel<<<EW(G * 256), 0, stream>>>(cellu, P(15), P(16), cf1, G, 9, 256, 1);
    launch_gemm(stream, cf1,   nullptr, Wlc1, P(18), cf2, G, 256, 256, 1);
    concat_gc_kernel<<<EW((size_t)G * 512), 0, stream>>>(gbuf, cf2, gc, G);
    launch_gemm(stream, gc,  nullptr, Woc0, P(46), cf1, G, 512, 256, 1);
    launch_gemm(stream, cf1, nullptr, Woc1, P(48), cf2, G, 256, 256, 1);
    launch_gemm(stream, cf2, nullptr, Woc2, P(50), o9,  G, 256, 9,   0);
    cell_final_kernel<<<EW(G * 9), 0, stream>>>(o9, cellu, outC, G);
#undef P
}
